// GCGRU_1151051235931
// MI455X (gfx1250) — matrix operand order repaired and checked by reading, not compiled, not run on hardware
//
#include <hip/hip_runtime.h>
#include <hip/hip_bf16.h>

typedef __bf16 bf16_t;
typedef __bf16 v16bf __attribute__((ext_vector_type(16)));
typedef float  v8f   __attribute__((ext_vector_type(8)));

#define NN   20000
#define TT   8
#define DD   256
#define D3   768          // 3 fused gate projections
#define DOUT 64
#define EE   320000

// ---------------------------------------------------------------------------
// Elementwise / graph-prep kernels
// ---------------------------------------------------------------------------
__global__ __launch_bounds__(256) void k_fill_f32(float* p, float v, int n) {
    int i = blockIdx.x * blockDim.x + threadIdx.x;
    if (i < n) p[i] = v;
}

__global__ __launch_bounds__(256) void k_accum_deg(const int* __restrict__ dst,
                                                   float* __restrict__ deg, int e) {
    int i = blockIdx.x * blockDim.x + threadIdx.x;
    if (i < e) atomicAdd(&deg[dst[i]], 1.0f);
}

__global__ __launch_bounds__(256) void k_norms(const float* __restrict__ deg,
                                               float* __restrict__ dis,
                                               float* __restrict__ snorm, int n) {
    int i = blockIdx.x * blockDim.x + threadIdx.x;
    if (i < n) {
        float d = deg[i];
        dis[i] = rsqrtf(d);
        snorm[i] = 1.0f / d;
    }
}

__global__ __launch_bounds__(256) void k_enorm(const int* __restrict__ src,
                                               const int* __restrict__ dst,
                                               const float* __restrict__ dis,
                                               float* __restrict__ enorm, int e) {
    int i = blockIdx.x * blockDim.x + threadIdx.x;
    if (i < e) enorm[i] = dis[src[i]] * dis[dst[i]];
}

__global__ __launch_bounds__(256) void k_f32_to_bf16(const float* __restrict__ in,
                                                     bf16_t* __restrict__ out, int n) {
    int i = blockIdx.x * blockDim.x + threadIdx.x;
    if (i < n) out[i] = (bf16_t)in[i];
}

// Pack three [256,256] f32 weights into one [256,768] bf16 matrix
__global__ __launch_bounds__(256) void k_pack_w3(const float* __restrict__ W0,
                                                 const float* __restrict__ W1,
                                                 const float* __restrict__ W2,
                                                 bf16_t* __restrict__ out) {
    int i = blockIdx.x * blockDim.x + threadIdx.x;
    if (i < DD * D3) {
        int k = i / D3, j = i % D3;
        int sel = j >> 8, jj = j & 255;
        const float* W = (sel == 0) ? W0 : (sel == 1) ? W1 : W2;
        out[i] = (bf16_t)W[k * DD + jj];
    }
}

// Pack three [256] f32 biases into one [768] f32 vector
__global__ __launch_bounds__(256) void k_pack_b3(const float* __restrict__ b0,
                                                 const float* __restrict__ b1,
                                                 const float* __restrict__ b2,
                                                 float* __restrict__ out) {
    int j = blockIdx.x * blockDim.x + threadIdx.x;
    if (j < D3) {
        int sel = j >> 8, jj = j & 255;
        out[j] = (sel == 0) ? b0[jj] : (sel == 1) ? b1[jj] : b2[jj];
    }
}

// x is [N, T, D]; extract timestep t as bf16 [N, D]
__global__ __launch_bounds__(256) void k_gather_xt(const float* __restrict__ x,
                                                   bf16_t* __restrict__ out, int t) {
    int i = blockIdx.x * blockDim.x + threadIdx.x;
    if (i < NN * DD) {
        int n = i / DD, d = i % DD;
        out[i] = (bf16_t)x[(size_t)n * (TT * DD) + (size_t)t * DD + d];
    }
}

// out[n,d] = vw[n,d] * snorm[n] + bias[d]   over [N, 768]
__global__ __launch_bounds__(256) void k_gcn_init(const float* __restrict__ vw,
                                                  const float* __restrict__ snorm,
                                                  const float* __restrict__ bias,
                                                  float* __restrict__ out) {
    int i = blockIdx.x * blockDim.x + threadIdx.x;
    if (i < NN * D3) {
        int n = i / D3, d = i % D3;
        out[i] = vw[i] * snorm[n] + bias[d];
    }
}

// out[dst] += enorm[e] * vw[src] over [N,768]; one thread per (edge, float4)
__global__ __launch_bounds__(256) void k_gcn_scatter(const int* __restrict__ src,
                                                     const int* __restrict__ dst,
                                                     const float* __restrict__ enorm,
                                                     const float* __restrict__ vw,
                                                     float* __restrict__ out) {
    long long i = (long long)blockIdx.x * blockDim.x + threadIdx.x;
    const int CPE = D3 / 4;                       // 192 chunks per edge
    if (i >= (long long)EE * CPE) return;
    int e = (int)(i / CPE);
    int c = (int)(i % CPE) * 4;
    float en = enorm[e];
    int s = src[e], d = dst[e];
    float4 v = *(const float4*)(vw + (size_t)s * D3 + c);
    float* o = out + (size_t)d * D3 + c;
    atomicAdd(o + 0, en * v.x);
    atomicAdd(o + 1, en * v.y);
    atomicAdd(o + 2, en * v.z);
    atomicAdd(o + 3, en * v.w);
}

// gcx/gch are [N,768]: cols 0-255 = r-gate, 256-511 = z-gate, 512-767 = n-gate
__global__ __launch_bounds__(256) void k_gru_update(const float* __restrict__ gcx,
                                                    const float* __restrict__ gch,
                                                    float* __restrict__ h) {
    int i = blockIdx.x * blockDim.x + threadIdx.x;
    if (i < NN * DD) {
        int n = i / DD, d = i % DD;
        size_t b = (size_t)n * D3 + d;
        float r  = 1.0f / (1.0f + expf(-(gcx[b] + gch[b])));
        float z  = 1.0f / (1.0f + expf(-(gcx[b + 256] + gch[b + 256])));
        float nv = tanhf(gcx[b + 512] + r * gch[b + 512]);
        h[i] = (1.0f - z) * h[i] + z * nv;
    }
}

// ---------------------------------------------------------------------------
// WMMA bf16 GEMM: C[M,Nc] = A[M,K] @ B[K,Nc] (+ optional bias[Nc])
// Block: 128 threads = 4 waves. Block tile 128x64; each wave computes 32x64
// (2 A-fragments x 4 B-fragments = 8 v_wmma_f32_16x16x32_bf16 per K-chunk).
// ---------------------------------------------------------------------------
#define TM 128
#define TN 64
#define TK 32

__global__ __launch_bounds__(128) void k_wmma_gemm(const bf16_t* __restrict__ A,
                                                   const bf16_t* __restrict__ B,
                                                   const float* __restrict__ bias,
                                                   float* __restrict__ C,
                                                   int M, int K, int Nc,
                                                   int has_bias) {
    __shared__ bf16_t lA[TM][TK];   // A tile, row-major          (8 KB)
    __shared__ bf16_t lBT[TN][TK];  // B tile, stored transposed  (4 KB)

    const int tid  = threadIdx.x;
    const int wave = tid >> 5;
    const int lane = tid & 31;
    const int half = lane >> 4;   // 0: lanes 0-15, 1: lanes 16-31
    const int l16  = lane & 15;

    const int m0 = blockIdx.x * TM;
    const int n0 = blockIdx.y * TN;

    v8f acc[2][4];
#pragma unroll
    for (int mi = 0; mi < 2; ++mi)
#pragma unroll
        for (int s = 0; s < 4; ++s)
#pragma unroll
            for (int j = 0; j < 8; ++j) acc[mi][s][j] = 0.0f;

    for (int k0 = 0; k0 < K; k0 += TK) {
        // Load A tile: 128x32 bf16 = 512 chunks of 8 bf16 (16B), coalesced.
#pragma unroll
        for (int c = tid; c < 512; c += 128) {
            int r  = c >> 2;              // 0..127
            int cc = (c & 3) << 3;        // 0,8,16,24
            int gr = m0 + r;
            uint4 v = make_uint4(0u, 0u, 0u, 0u);
            if (gr < M) v = *(const uint4*)(A + (size_t)gr * K + k0 + cc);
            *(uint4*)(&lA[r][cc]) = v;
        }
        // Load B tile 32x64, coalesced 16B global reads, transposed LDS store.
#pragma unroll
        for (int c = tid; c < 256; c += 128) {
            int kk = c >> 3;              // 0..31
            int cg = (c & 7) << 3;        // 0..56
            uint4 v = *(const uint4*)(B + (size_t)(k0 + kk) * Nc + n0 + cg);
            const bf16_t* pv = (const bf16_t*)&v;
#pragma unroll
            for (int j = 0; j < 8; ++j) lBT[cg + j][kk] = pv[j];
        }
        // Prefetch next A K-slice (global_prefetch_b8), one row per thread.
        if (k0 + TK < K && m0 + tid < M)
            __builtin_prefetch(A + (size_t)(m0 + tid) * K + k0 + TK, 0, 3);
        __syncthreads();

        // A fragments (16x32 each): row = l16, K(e) = 16*(e/8) + 8*half + e%8
        v16bf afrag[2];
#pragma unroll
        for (int mi = 0; mi < 2; ++mi) {
            const bf16_t* pr = &lA[wave * 32 + mi * 16 + l16][0];
            ((uint4*)&afrag[mi])[0] = *(const uint4*)(pr + 8 * half);
            ((uint4*)&afrag[mi])[1] = *(const uint4*)(pr + 16 + 8 * half);
        }
#pragma unroll
        for (int s = 0; s < 4; ++s) {
            // B fragment (32x16): col = l16, K(e) = 16*(e/8) + 8*half + e%8
            v16bf bfrag;
            const bf16_t* pc = &lBT[s * 16 + l16][0];
            ((uint4*)&bfrag)[0] = *(const uint4*)(pc + 8 * half);
            ((uint4*)&bfrag)[1] = *(const uint4*)(pc + 16 + 8 * half);
#pragma unroll
            for (int mi = 0; mi < 2; ++mi)
                acc[mi][s] = __builtin_amdgcn_wmma_f32_16x16x32_bf16(
                    false, afrag[mi], false, bfrag, (short)0, acc[mi][s], false, false);
        }
        __syncthreads();
    }

    // C/D layout: VGPR i -> row = i + 8*half, col = l16
#pragma unroll
    for (int mi = 0; mi < 2; ++mi)
#pragma unroll
        for (int s = 0; s < 4; ++s) {
            int col = n0 + s * 16 + l16;
            float bv = has_bias ? bias[col] : 0.0f;
#pragma unroll
            for (int i = 0; i < 8; ++i) {
                int row = m0 + wave * 32 + mi * 16 + i + 8 * half;
                if (row < M) C[(size_t)row * Nc + col] = acc[mi][s][i] + bv;
            }
        }
}

// ---------------------------------------------------------------------------
// Host-side orchestration
// ---------------------------------------------------------------------------
static inline int ceil_div(int a, int b) { return (a + b - 1) / b; }

extern "C" void kernel_launch(void* const* d_in, const int* in_sizes, int n_in,
                              void* d_out, int out_size, void* d_ws, size_t ws_size,
                              hipStream_t stream) {
    (void)in_sizes; (void)n_in; (void)out_size; (void)ws_size;

    const float* x    = (const float*)d_in[0];
    const int*   edge = (const int*)d_in[1];
    const float* W_xr = (const float*)d_in[2];  const float* b_xr = (const float*)d_in[3];
    const float* W_hr = (const float*)d_in[4];  const float* b_hr = (const float*)d_in[5];
    const float* W_xz = (const float*)d_in[6];  const float* b_xz = (const float*)d_in[7];
    const float* W_hz = (const float*)d_in[8];  const float* b_hz = (const float*)d_in[9];
    const float* W_hn = (const float*)d_in[10]; const float* b_hn = (const float*)d_in[11];
    const float* W_fc = (const float*)d_in[12]; const float* b_fc = (const float*)d_in[13];
    float* out = (float*)d_out;

    const int* srcIdx = edge;
    const int* dstIdx = edge + EE;

    // ---- workspace carve-out ----
    char* wsp = (char*)d_ws;
    auto alloc = [&](size_t bytes) -> char* {
        char* p = wsp;
        wsp += (bytes + 255) & ~(size_t)255;
        return p;
    };
    float*  deg   = (float*)alloc((size_t)NN * 4);
    float*  dis   = (float*)alloc((size_t)NN * 4);
    float*  snorm = (float*)alloc((size_t)NN * 4);
    float*  enorm = (float*)alloc((size_t)EE * 4);
    float*  h     = (float*)alloc((size_t)NN * DD * 4);
    bf16_t* x_bf  = (bf16_t*)alloc((size_t)NN * DD * 2);
    bf16_t* h_bf  = (bf16_t*)alloc((size_t)NN * DD * 2);
    bf16_t* Wx_c  = (bf16_t*)alloc((size_t)DD * D3 * 2);   // [Wxr|Wxz|Whn]
    bf16_t* Wh_c  = (bf16_t*)alloc((size_t)DD * D3 * 2);   // [Whr|Whz|Whn]
    bf16_t* Wfc_b = (bf16_t*)alloc((size_t)DD * DOUT * 2);
    float*  bx_c  = (float*)alloc((size_t)D3 * 4);          // [bxr|bxz|bhn]
    float*  bh_c  = (float*)alloc((size_t)D3 * 4);          // [bhr|bhz|bhn]
    float*  vwx   = (float*)alloc((size_t)NN * D3 * 4);     // x projections
    float*  vwh   = (float*)alloc((size_t)NN * D3 * 4);     // h projections
    float*  gcx   = (float*)alloc((size_t)NN * D3 * 4);     // gcn(x) results
    float*  gch   = (float*)alloc((size_t)NN * D3 * 4);     // gcn(h) results

    const int ND  = NN * DD;
    const int ND3 = NN * D3;
    const dim3 b256(256);
    const dim3 gN(ceil_div(NN, 256));
    const dim3 gE(ceil_div(EE, 256));
    const dim3 gND(ceil_div(ND, 256));
    const dim3 gND3(ceil_div(ND3, 256));
    const dim3 gScatter(ceil_div(EE * (D3 / 4), 256));
    const dim3 gemmBlk(128);
    const dim3 gemmGrid(ceil_div(NN, TM), D3 / TN);      // 157 x 12
    const dim3 gemmGridFc(ceil_div(NN, TM), DOUT / TN);  // 157 x 1

    // ---- graph normalization precompute ----
    k_fill_f32<<<gN, b256, 0, stream>>>(deg, 1.0f, NN);   // self loop
    k_accum_deg<<<gE, b256, 0, stream>>>(dstIdx, deg, EE);
    k_norms<<<gN, b256, 0, stream>>>(deg, dis, snorm, NN);
    k_enorm<<<gE, b256, 0, stream>>>(srcIdx, dstIdx, dis, enorm, EE);

    // ---- pack fused weights / biases, convert to bf16 ----
    const dim3 gW3(ceil_div(DD * D3, 256));
    k_pack_w3<<<gW3, b256, 0, stream>>>(W_xr, W_xz, W_hn, Wx_c);
    k_pack_w3<<<gW3, b256, 0, stream>>>(W_hr, W_hz, W_hn, Wh_c);
    k_pack_b3<<<dim3(ceil_div(D3, 256)), b256, 0, stream>>>(b_xr, b_xz, b_hn, bx_c);
    k_pack_b3<<<dim3(ceil_div(D3, 256)), b256, 0, stream>>>(b_hr, b_hz, b_hn, bh_c);
    k_f32_to_bf16<<<dim3(ceil_div(DD * DOUT, 256)), b256, 0, stream>>>(W_fc, Wfc_b, DD * DOUT);

    // ---- init hidden state ----
    k_fill_f32<<<gND, b256, 0, stream>>>(h, 0.0f, ND);

    for (int t = 0; t < TT; ++t) {
        k_gather_xt<<<gND, b256, 0, stream>>>(x, x_bf, t);
        k_f32_to_bf16<<<gND, b256, 0, stream>>>(h, h_bf, ND);

        // fused projections: vwx = x_t @ [Wxr|Wxz|Whn], vwh = h @ [Whr|Whz|Whn]
        k_wmma_gemm<<<gemmGrid, gemmBlk, 0, stream>>>(x_bf, Wx_c, nullptr, vwx, NN, DD, D3, 0);
        k_wmma_gemm<<<gemmGrid, gemmBlk, 0, stream>>>(h_bf, Wh_c, nullptr, vwh, NN, DD, D3, 0);

        // gcn: self-loop + bias init, then edge scatter-aggregate
        k_gcn_init<<<gND3, b256, 0, stream>>>(vwx, snorm, bx_c, gcx);
        k_gcn_init<<<gND3, b256, 0, stream>>>(vwh, snorm, bh_c, gch);
        k_gcn_scatter<<<gScatter, b256, 0, stream>>>(srcIdx, dstIdx, enorm, vwx, gcx);
        k_gcn_scatter<<<gScatter, b256, 0, stream>>>(srcIdx, dstIdx, enorm, vwh, gch);

        k_gru_update<<<gND, b256, 0, stream>>>(gcx, gch, h);
    }

    // ---- final projection: out = h @ W_fc + b_fc ----
    k_f32_to_bf16<<<gND, b256, 0, stream>>>(h, h_bf, ND);
    k_wmma_gemm<<<gemmGridFc, gemmBlk, 0, stream>>>(h_bf, Wfc_b, b_fc, out, NN, DD, DOUT, 1);
}
